// WeightedConv2D_56289841381897
// MI455X (gfx1250) — matrix operand order repaired and checked by reading, not compiled, not run on hardware
//
#include <hip/hip_runtime.h>
#include <math.h>

typedef _Float16 f16;
typedef _Float16 __attribute__((ext_vector_type(16))) v16h;
typedef _Float16 __attribute__((ext_vector_type(8)))  v8h;
typedef float    __attribute__((ext_vector_type(8)))  v8f;

#define BN    4
#define CCH   64
#define OCH   64
#define NNPTS 16384
#define KW    9
#define PADW  4
#define SIGMA 0.08f
#define EPSV  1e-5f

// workspace layout (bytes)
#define OFF_G   ((size_t)0)                      // int g[4][B][N]          1 MB
#define OFF_WF  ((size_t)0x100000)               // f16 conv frags 147456
#define OFF_FF  ((size_t)0x150000)               // f16 fusion frags 16384
#define OFF_ST  ((size_t)0x158000)               // float stats[128]
#define OFF_Y   ((size_t)0x160000)               // f16 Y[4][B][64][N]      32 MB
#define OFF_Z   (OFF_Y + (size_t)4*BN*OCH*NNPTS*2) // float Z[B][64][N]     16 MB

// ---------------------------------------------------------------------------
// Compose the compounding gathers: g0 = idx0 ; g_i[n] = g_{i-1}[idx_i[n]]
__global__ void prep_idx(const int* __restrict__ indices, int* __restrict__ g) {
  const int b = blockIdx.x;
  for (int n = threadIdx.x; n < NNPTS; n += blockDim.x)
    g[(size_t)b * NNPTS + n] = indices[((size_t)b * NNPTS + n) * 4 + 0];
  __threadfence();
  __syncthreads();
  for (int i = 1; i < 4; ++i) {
    for (int n = threadIdx.x; n < NNPTS; n += blockDim.x) {
      int t = indices[((size_t)b * NNPTS + n) * 4 + i];
      g[((size_t)i * BN + b) * NNPTS + n] = g[((size_t)(i - 1) * BN + b) * NNPTS + t];
    }
    __threadfence();
    __syncthreads();
  }
}

// ---------------------------------------------------------------------------
// A-fragment k order used here (16x32, wave32; contiguous, B uses the same):
// lanes 0-15 hold K{0..15}, lanes 16-31 hold K{16..31}, 2 per VGPR.
__device__ __forceinline__ int a_layout_kk(int lane, int e) {
  int vg = e >> 1;
  int grp = vg >> 2;
  int half = (lane >> 4) & 1;
  (void)grp;
  return half * 16 + e;
}

// K-dim flattening for the conv GEMM is kidx = k*64 + c  (so each 32-chunk has
// constant k and contiguous c). Fusion uses kidx = i*64 + o directly.
__global__ void pack_weights(const float* __restrict__ convw,
                             const float* __restrict__ fusw,
                             f16* __restrict__ wf, f16* __restrict__ ff) {
  const int T1 = 4 * 4 * 18 * 512;          // curves * o-tiles * k-chunks * 512
  const int T2 = 4 * 8 * 512;               // o'-tiles * k-chunks * 512
  int gid = blockIdx.x * blockDim.x + threadIdx.x;
  if (gid < T1) {
    int e = gid & 15, lane = (gid >> 4) & 31;
    int q = (gid >> 9) % 18;
    int t = ((gid >> 9) / 18) % 4;
    int i = (gid >> 9) / 72;
    int kk = a_layout_kk(lane, e);
    int kidx = q * 32 + kk;                  // < 576
    int c = kidx & 63, k = kidx >> 6;        // flatten order: k*64 + c
    int o = t * 16 + (lane & 15);
    wf[gid] = (f16)convw[(((size_t)i * OCH + o) * CCH + c) * KW + k];
  } else if (gid < T1 + T2) {
    int g2 = gid - T1;
    int e = g2 & 15, lane = (g2 >> 4) & 31;
    int q = (g2 >> 9) & 7;
    int t = g2 >> 12;
    int kk = a_layout_kk(lane, e);
    int kidx = q * 32 + kk;                  // < 256
    int o = t * 16 + (lane & 15);
    ff[g2] = (f16)fusw[o * 256 + kidx];
  }
}

__global__ void zero_stats(float* __restrict__ stats) {
  if (threadIdx.x < 128) stats[threadIdx.x] = 0.0f;
}

// ---------------------------------------------------------------------------
// Weighted conv per (curve, batch, 128-position block).
// 4 waves; each wave: 2 n-subtiles x 4 o-tiles, 18 K-chunks of WMMA f16.
// A fragments: one per-lane base pointer + constant immediate offsets.
// B fragment per chunk = 2x ds_load_b128 scaled by one broadcast dw value.
__global__ __launch_bounds__(128) void conv_kernel(
    const float* __restrict__ input, const float* __restrict__ mc,
    const int* __restrict__ g, const f16* __restrict__ wf,
    f16* __restrict__ Y) {
  __shared__ f16   xgT[136][72];   // [tile pos j][channel c], pitch 72 for banks
  __shared__ float ct[3][136];     // coords tile + halo
  __shared__ f16   dwh[KW][128];   // window weights
  __shared__ int   gj[136];        // composed gather index per tile position

  const int n0 = blockIdx.x * 128;
  const int b = blockIdx.y;
  const int i = blockIdx.z;
  const int tid = threadIdx.x;

  for (int j = tid; j < 136; j += 128) {
    int m = n0 + j - PADW;
    bool inr = (m >= 0) && (m < NNPTS);
    gj[j] = inr ? g[((size_t)i * BN + b) * NNPTS + m] : 0;
#pragma unroll
    for (int d = 0; d < 3; ++d)
      ct[d][j] = inr ? mc[(((size_t)b * 3 + d) * NNPTS + m) * 4 + i] : 0.0f;
  }
  __syncthreads();

  {  // gathered input tile -> LDS f16, transposed
    int c = tid & 63;
    int jt = tid >> 6;
    const float* inb = input + ((size_t)b * CCH + c) * NNPTS;
    for (int jj = 0; jj < 68; ++jj) {
      int j = jt * 68 + jj;
      int m = n0 + j - PADW;
      float v = 0.0f;
      if (m >= 0 && m < NNPTS) v = inb[gj[j]];
      xgT[j][c] = (f16)v;
    }
  }
  for (int e3 = tid; e3 < KW * 128; e3 += 128) {
    int k = e3 >> 7;
    int j = e3 & 127;
    float dx = ct[0][j + k] - ct[0][j + PADW];
    float dy = ct[1][j + k] - ct[1][j + PADW];
    float dz = ct[2][j + k] - ct[2][j + PADW];
    float w = 1.0f - sqrtf(dx * dx + dy * dy + dz * dz) * (1.0f / SIGMA);
    dwh[k][j] = (f16)fmaxf(w, 0.0f);
  }
  __syncthreads();

  const int wave = tid >> 5;
  const int lane = tid & 31;
  const int hi16 = lane & 16;
  const int col = lane & 15;

  v8f zero = {};
  v8f acc[2][4];
#pragma unroll
  for (int s = 0; s < 2; ++s)
#pragma unroll
    for (int t = 0; t < 4; ++t) acc[s][t] = zero;

  // one flat byte pointer; all fragment loads use constant immediate offsets
  const char* ap = (const char*)wf + ((size_t)(i * 4 * 18) * 32 + (size_t)lane) * 32;

#pragma unroll
  for (int k = 0; k < 9; ++k) {
    f16 dv[2];
#pragma unroll
    for (int s = 0; s < 2; ++s)
      dv[s] = dwh[k][wave * 32 + s * 16 + col];
#pragma unroll
    for (int h = 0; h < 2; ++h) {
      const int q = 2 * k + h;
      v16h a0 = *(const v16h*)(ap + (0 * 18 + q) * 1024);
      v16h a1 = *(const v16h*)(ap + (1 * 18 + q) * 1024);
      v16h a2 = *(const v16h*)(ap + (2 * 18 + q) * 1024);
      v16h a3 = *(const v16h*)(ap + (3 * 18 + q) * 1024);
      const int c0 = h * 32 + hi16;          // contiguous channel run start
#pragma unroll
      for (int s = 0; s < 2; ++s) {
        const int jcol = wave * 32 + s * 16 + col;
        const f16 dval = dv[s];
        const v8h xlo = *(const v8h*)&xgT[jcol + k][c0];
        const v8h xhi = *(const v8h*)&xgT[jcol + k][c0 + 8];
        v16h bm;
#pragma unroll
        for (int e = 0; e < 8; ++e) {
          bm[e] = xlo[e] * dval;
          bm[e + 8] = xhi[e] * dval;
        }
        acc[s][0] = __builtin_amdgcn_wmma_f32_16x16x32_f16(false, a0, false, bm, (short)0, acc[s][0], false, false);
        acc[s][1] = __builtin_amdgcn_wmma_f32_16x16x32_f16(false, a1, false, bm, (short)0, acc[s][1], false, false);
        acc[s][2] = __builtin_amdgcn_wmma_f32_16x16x32_f16(false, a2, false, bm, (short)0, acc[s][2], false, false);
        acc[s][3] = __builtin_amdgcn_wmma_f32_16x16x32_f16(false, a3, false, bm, (short)0, acc[s][3], false, false);
      }
    }
  }

  f16* Yb = Y + ((size_t)i * BN + b) * OCH * NNPTS;
  const int obase = hi16 ? 8 : 0;
#pragma unroll
  for (int s = 0; s < 2; ++s) {
    int n = n0 + wave * 32 + s * 16 + col;
    char* yp = (char*)(Yb + (size_t)obase * NNPTS + n);   // + o*2*NNPTS via imm
#pragma unroll
    for (int t = 0; t < 4; ++t)
#pragma unroll
      for (int j = 0; j < 8; ++j)
        *(f16*)(yp + (size_t)(t * 16 + j) * (NNPTS * 2)) = (f16)acc[s][t][j];
  }
}

// ---------------------------------------------------------------------------
// Fusion (1x1 conv 256->64) on reindexed y, + bias, + per-channel sum/sumsq.
__global__ __launch_bounds__(128) void fusion_kernel(
    const f16* __restrict__ Y, const f16* __restrict__ ff,
    const int* __restrict__ reidx, const float* __restrict__ fusb,
    float* __restrict__ Z, float* __restrict__ stats) {
  __shared__ f16 cat2[64][264];      // [column j][cat channel], pitch 264
  __shared__ int re4[4][64];
  __shared__ float lsum[64], lsq[64];

  const int n0 = blockIdx.x * 64;
  const int b = blockIdx.y;
  const int tid = threadIdx.x;

  if (tid < 64) { lsum[tid] = 0.0f; lsq[tid] = 0.0f; }
  for (int e = tid; e < 256; e += 128) {
    int i = e >> 6, j = e & 63;
    re4[i][j] = reidx[((size_t)b * NNPTS + (n0 + j)) * 4 + i];
  }
  __syncthreads();

  for (int e2 = tid; e2 < 64 * 256; e2 += 128) {
    int j = e2 >> 8, cc = e2 & 255;
    int i = cc >> 6, o = cc & 63;
    int r = re4[i][j];
    cat2[j][cc] = Y[(((size_t)i * BN + b) * OCH + o) * NNPTS + r];
  }
  __syncthreads();

  const int wave = tid >> 5;
  const int lane = tid & 31;
  const int rowoff = lane & 16;
  const int col = lane & 15;
  const int jcol = wave * 16 + col;

  v8f zero = {};
  v8f acc[4] = {zero, zero, zero, zero};
  const char* fp = (const char*)ff + (size_t)lane * 32;
#pragma unroll
  for (int q = 0; q < 8; ++q) {
    v16h a0 = *(const v16h*)(fp + (0 * 8 + q) * 1024);
    v16h a1 = *(const v16h*)(fp + (1 * 8 + q) * 1024);
    v16h a2 = *(const v16h*)(fp + (2 * 8 + q) * 1024);
    v16h a3 = *(const v16h*)(fp + (3 * 8 + q) * 1024);
    const int co = q * 32 + rowoff;
    const v8h xlo = *(const v8h*)&cat2[jcol][co];
    const v8h xhi = *(const v8h*)&cat2[jcol][co + 8];
    v16h bm;
#pragma unroll
    for (int e = 0; e < 8; ++e) {
      bm[e] = xlo[e];
      bm[e + 8] = xhi[e];
    }
    acc[0] = __builtin_amdgcn_wmma_f32_16x16x32_f16(false, a0, false, bm, (short)0, acc[0], false, false);
    acc[1] = __builtin_amdgcn_wmma_f32_16x16x32_f16(false, a1, false, bm, (short)0, acc[1], false, false);
    acc[2] = __builtin_amdgcn_wmma_f32_16x16x32_f16(false, a2, false, bm, (short)0, acc[2], false, false);
    acc[3] = __builtin_amdgcn_wmma_f32_16x16x32_f16(false, a3, false, bm, (short)0, acc[3], false, false);
  }

  const int n = n0 + jcol;
  const int obase = rowoff ? 8 : 0;
  char* zp = (char*)(Z + ((size_t)b * OCH + obase) * NNPTS + n);
#pragma unroll
  for (int t = 0; t < 4; ++t) {
#pragma unroll
    for (int j = 0; j < 8; ++j) {
      int o = t * 16 + j + obase;
      float z = acc[t][j] + fusb[o];
      *(float*)(zp + (size_t)(t * 16 + j) * (NNPTS * 4)) = z;
      float s1 = z, s2 = z * z;
#pragma unroll
      for (int m = 1; m < 16; m <<= 1) {   // reduce across the 16 lanes sharing o
        s1 += __shfl_xor(s1, m, 32);
        s2 += __shfl_xor(s2, m, 32);
      }
      if (col == 0) { atomicAdd(&lsum[o], s1); atomicAdd(&lsq[o], s2); }
    }
  }
  __syncthreads();
  if (tid < 64) {
    atomicAdd(&stats[tid], lsum[tid]);
    atomicAdd(&stats[64 + tid], lsq[tid]);
  }
}

// ---------------------------------------------------------------------------
// BatchNorm (training stats, pads contribute z = bias) + ReLU.
__global__ void bn_kernel(const float* __restrict__ Z, const float* __restrict__ stats,
                          const float* __restrict__ fusb, const float* __restrict__ gamma,
                          const float* __restrict__ beta, float* __restrict__ out, int total) {
  int gid = blockIdx.x * blockDim.x + threadIdx.x;
  if (gid >= total) return;
  const int NP = NNPTS + 2;
  int n = gid % NP;
  int o = (gid / NP) % OCH;
  int b = gid / (NP * OCH);
  float bias = fusb[o];
  float cnt = (float)(BN * NP);
  float sum = stats[o] + 2.0f * BN * bias;
  float sq = stats[64 + o] + 2.0f * BN * bias * bias;
  float mean = sum / cnt;
  float var = sq / cnt - mean * mean;
  float z = (n >= 1 && n <= NNPTS) ? Z[((size_t)b * OCH + o) * NNPTS + (n - 1)] : bias;
  float r = (z - mean) * rsqrtf(var + EPSV) * gamma[o] + beta[o];
  out[gid] = fmaxf(r, 0.0f);
}

// ---------------------------------------------------------------------------
extern "C" void kernel_launch(void* const* d_in, const int* in_sizes, int n_in,
                              void* d_out, int out_size, void* d_ws, size_t ws_size,
                              hipStream_t stream) {
  (void)in_sizes; (void)n_in; (void)ws_size; (void)out_size;
  const float* input = (const float*)d_in[0];
  const float* mc = (const float*)d_in[1];
  const int* indices = (const int*)d_in[2];
  const int* reidx = (const int*)d_in[3];
  const float* convw = (const float*)d_in[4];
  const float* fusw = (const float*)d_in[5];
  const float* fusb = (const float*)d_in[6];
  const float* gamma = (const float*)d_in[7];
  const float* beta = (const float*)d_in[8];
  float* out = (float*)d_out;

  char* ws = (char*)d_ws;
  int* g = (int*)(ws + OFF_G);
  f16* wf = (f16*)(ws + OFF_WF);
  f16* ffr = (f16*)(ws + OFF_FF);
  float* stats = (float*)(ws + OFF_ST);
  f16* Y = (f16*)(ws + OFF_Y);
  float* Z = (float*)(ws + OFF_Z);

  prep_idx<<<BN, 1024, 0, stream>>>(indices, g);
  pack_weights<<<(4 * 4 * 18 * 512 + 4 * 8 * 512 + 255) / 256, 256, 0, stream>>>(convw, fusw, wf, ffr);
  zero_stats<<<1, 128, 0, stream>>>(stats);
  conv_kernel<<<dim3(NNPTS / 128, BN, 4), 128, 0, stream>>>(input, mc, g, wf, Y);
  fusion_kernel<<<dim3(NNPTS / 64, BN), 128, 0, stream>>>(Y, ffr, reidx, fusb, Z, stats);
  int total = BN * OCH * (NNPTS + 2);
  bn_kernel<<<(total + 255) / 256, 256, 0, stream>>>(Z, stats, fusb, gamma, beta, out, total);
}
